// SelfAttention_13915694039152
// MI455X (gfx1250) — matrix operand order repaired and checked by reading, not compiled, not run on hardware
//
#include <hip/hip_runtime.h>
#include <hip/hip_bf16.h>

// Problem constants (from reference): b=4, t=1024, d=1024, heads=16, hd=64
#define B_  4
#define T_  1024
#define D_  1024
#define H_  16
#define HD_ 64
#define QT_ 16   // q rows per wave tile
#define KC_ 32   // key positions per chunk

typedef __bf16 bf16;
typedef __attribute__((ext_vector_type(16))) __bf16 v16bf;
typedef __attribute__((ext_vector_type(8)))  float  v8f;
typedef __attribute__((ext_vector_type(4)))  unsigned int v4u;
typedef __attribute__((ext_vector_type(8)))  int    v8i;
typedef __attribute__((ext_vector_type(4)))  int    v4i;

union BF16Frag {
  v16bf v;
  unsigned int u[8];
  bf16 h[16];
};

__device__ __forceinline__ v8f wmma_bf16(const BF16Frag& a, const BF16Frag& b, v8f c) {
  // D = A(16x32 bf16) * B(32x16 bf16) + C(16x16 f32)
  return __builtin_amdgcn_wmma_f32_16x16x32_bf16(
      /*neg_a=*/false, a.v, /*neg_b=*/false, b.v,
      /*c_mod=*/(short)0, c, /*reuse_a=*/false, /*reuse_b=*/false);
}

// Load bf16 p[0..7] and p[16..23] -> B-fragment lane data (two b128 loads)
__device__ __forceinline__ void load16(BF16Frag& f, const bf16* p) {
  const uint4* q = reinterpret_cast<const uint4*>(p);
  uint4 a = q[0], b = q[2];
  f.u[0]=a.x; f.u[1]=a.y; f.u[2]=a.z; f.u[3]=a.w;
  f.u[4]=b.x; f.u[5]=b.y; f.u[6]=b.z; f.u[7]=b.w;
}

// A-fragment lane data: K = koff+{0..7} in VGPR0..3, K = koff+{16..23} in VGPR4..7
__device__ __forceinline__ void loadA(BF16Frag& f, const bf16* row, int koff) {
  const uint4* q0 = reinterpret_cast<const uint4*>(row + koff);
  const uint4* q1 = reinterpret_cast<const uint4*>(row + koff + 16);
  uint4 a = *q0, b = *q1;
  f.u[0]=a.x; f.u[1]=a.y; f.u[2]=a.z; f.u[3]=a.w;
  f.u[4]=b.x; f.u[5]=b.y; f.u[6]=b.z; f.u[7]=b.w;
}

// ---------------------------------------------------------------- cvt f32->bf16
__global__ void __launch_bounds__(256) cvt_f32_bf16(const float* __restrict__ in,
                                                    bf16* __restrict__ out, int n) {
  int i = blockIdx.x * blockDim.x + threadIdx.x;
  if (i < n) out[i] = (bf16)in[i];
}

// ---------------------------------------------------------------- GEMM: A[M,K] * B[N,K]^T
// Wave tile: 32(M) x 64(N) = 8 accumulator tiles; double-buffered fragments so
// global_load_b128 of step k+32 overlap the WMMAs of step k.
// Block = 8 waves = 256(M) x 64(N).
__global__ void __launch_bounds__(256) gemm_bf16_nt(const bf16* __restrict__ A,
                                                    const bf16* __restrict__ Bm,
                                                    int M, int N, int K,
                                                    float* __restrict__ outF,
                                                    bf16* __restrict__ outB,
                                                    const unsigned char* __restrict__ mask) {
  const int lane = threadIdx.x & 31;
  const int wave = threadIdx.x >> 5;
  const int grp  = lane >> 4;   // 0 | 1 (lane group)
  const int ln   = lane & 15;
  const int m0 = blockIdx.y * 256 + wave * 32;
  const int n0 = blockIdx.x * 64;

  v8f acc[2][4] = {};
  const bf16* arow0 = A + (size_t)(m0 + ln) * K;
  const bf16* arow1 = A + (size_t)(m0 + 16 + ln) * K;
  const bf16* brow[4];
#pragma unroll
  for (int j = 0; j < 4; ++j) brow[j] = Bm + (size_t)(n0 + j * 16 + ln) * K + grp * 8;

  BF16Frag a[2][2], b[2][4];
  // preload buffer 0 @ k = 0
  loadA(a[0][0], arow0, grp * 8);
  loadA(a[0][1], arow1, grp * 8);
#pragma unroll
  for (int j = 0; j < 4; ++j) load16(b[0][j], brow[j]);

  int k = 0;
  for (; k + 64 < K; k += 64) {
    // prefetch buf1 @ k+32
    loadA(a[1][0], arow0, k + 32 + grp * 8);
    loadA(a[1][1], arow1, k + 32 + grp * 8);
#pragma unroll
    for (int j = 0; j < 4; ++j) load16(b[1][j], brow[j] + k + 32);
    // multiply buf0 @ k
#pragma unroll
    for (int j = 0; j < 4; ++j) {
      acc[0][j] = wmma_bf16(a[0][0], b[0][j], acc[0][j]);
      acc[1][j] = wmma_bf16(a[0][1], b[0][j], acc[1][j]);
    }
    // prefetch buf0 @ k+64
    loadA(a[0][0], arow0, k + 64 + grp * 8);
    loadA(a[0][1], arow1, k + 64 + grp * 8);
#pragma unroll
    for (int j = 0; j < 4; ++j) load16(b[0][j], brow[j] + k + 64);
    // multiply buf1 @ k+32
#pragma unroll
    for (int j = 0; j < 4; ++j) {
      acc[0][j] = wmma_bf16(a[1][0], b[1][j], acc[0][j]);
      acc[1][j] = wmma_bf16(a[1][1], b[1][j], acc[1][j]);
    }
  }
  // tail: k = K-64
  loadA(a[1][0], arow0, k + 32 + grp * 8);
  loadA(a[1][1], arow1, k + 32 + grp * 8);
#pragma unroll
  for (int j = 0; j < 4; ++j) load16(b[1][j], brow[j] + k + 32);
#pragma unroll
  for (int j = 0; j < 4; ++j) {
    acc[0][j] = wmma_bf16(a[0][0], b[0][j], acc[0][j]);
    acc[1][j] = wmma_bf16(a[0][1], b[0][j], acc[1][j]);
  }
#pragma unroll
  for (int j = 0; j < 4; ++j) {
    acc[0][j] = wmma_bf16(a[1][0], b[1][j], acc[0][j]);
    acc[1][j] = wmma_bf16(a[1][1], b[1][j], acc[1][j]);
  }

#pragma unroll
  for (int i = 0; i < 2; ++i) {
#pragma unroll
    for (int j = 0; j < 4; ++j) {
#pragma unroll
      for (int v = 0; v < 8; ++v) {
        const int m = m0 + i * 16 + v + grp * 8;  // C layout: row = vgpr + 8*(lane>=16)
        const int n = n0 + j * 16 + ln;           //           col = lane % 16
        float val = acc[i][j][v];
        if (outB) {
          outB[(size_t)m * N + n] = (bf16)val;
        } else {
          if (mask && !mask[m]) val = 0.0f;       // fused TensorMask.apply_mask
          outF[(size_t)m * N + n] = val;
        }
      }
    }
  }
}

// ---------------------------------------------------------------- flash attention
// One wave (32 threads) per (b, h, q-tile of 16 rows). Key chunks of 32.
// V chunk staging uses the Tensor Data Mover (2D tile: 32 rows x 64 bf16,
// row stride 3*D elements), tracked with TENSORcnt.
__global__ void __launch_bounds__(32) attn_kernel(const bf16* __restrict__ qkv,
                                                  const unsigned char* __restrict__ mask,
                                                  bf16* __restrict__ ctx) {
  __shared__ __align__(16) bf16 sV[KC_][HD_];  // staged V chunk (32 keys x 64 feats)
  __shared__ __align__(16) bf16 sP[QT_][KC_];  // probs, C-layout -> A-layout round trip

  const int lane = threadIdx.x;
  const int grp  = lane >> 4;
  const int ln   = lane & 15;
  const int qt = blockIdx.x, h = blockIdx.y, b = blockIdx.z;
  const int qbase = qt * QT_;
  const float scale = 0.125f;  // 1/sqrt(64)
  const int QKVD = 3 * D_;

  // LDS byte offset of sV (LDS aperture: addr[31:0] == LDS offset)
  const unsigned sV_lds = (unsigned)(size_t)(&sV[0][0]);

  // Q A-fragments: feats 0..31 (qa0) and 32..63 (qa1); rows = qbase+ln, K-contiguous.
  BF16Frag qa0, qa1;
  {
    const bf16* qp = qkv + (size_t)(b * T_ + qbase + ln) * QKVD + h * HD_ + grp * 8;
    loadA(qa0, qp, 0);
    loadA(qa1, qp + 32, 0);
  }

  bool mq[8];
#pragma unroll
  for (int v = 0; v < 8; ++v) mq[v] = mask[b * T_ + qbase + v + grp * 8] != 0;

  v8f o[4] = {};
  float mrow[8], lrow[8];
#pragma unroll
  for (int v = 0; v < 8; ++v) { mrow[v] = -1e30f; lrow[v] = 0.0f; }

  const int nch = (qbase + QT_ - 1) / KC_ + 1;  // causal chunk limit
  for (int kc = 0; kc < nch; ++kc) {
    const int k0 = kc * KC_;

    // ---- TDM: stage V chunk into sV (async, TENSORcnt) ----
    {
      const bf16* vbase = qkv + (size_t)(b * T_ + k0) * QKVD + 2 * D_ + h * HD_;
      const unsigned long long ga = (unsigned long long)(size_t)vbase;
      // D# group 0: count=1 | lds_addr | global_addr(57b) | type=2
      v4u g0 = { 1u,
                 sV_lds,
                 (unsigned)(ga & 0xffffffffull),
                 (unsigned)((ga >> 32) & 0x01ffffffull) | 0x80000000u };
      // D# group 1: data_size=2B; tensor_dim0=64, tensor_dim1=32;
      //             tile_dim0=64, tile_dim1=32; tensor_dim0_stride=3*D
      v8i g1 = { (int)0x00010000u,        // wg_mask=0, data_size=1 (2 bytes)
                 (int)(64u << 16),        // tensor_dim0[15:0] << 16
                 (int)(32u << 16),        // tensor_dim1[15:0] << 16
                 (int)(64u << 16),        // tile_dim0 << 16
                 (int)32,                 // tile_dim1
                 (int)(3 * D_),           // tensor_dim0_stride[31:0]
                 0, 0 };
      v4i gz = { 0, 0, 0, 0 };
#if __clang_major__ >= 23
      v8i gz8 = { 0, 0, 0, 0, 0, 0, 0, 0 };
      __builtin_amdgcn_tensor_load_to_lds(g0, g1, gz, gz, gz8, 0);
#else
      __builtin_amdgcn_tensor_load_to_lds(g0, g1, gz, gz, 0);
#endif
    }

    // S = Q K^T for 2 key sub-tiles; K rows are feat-contiguous -> direct B-frag loads.
    BF16Frag kb00, kb01, kb10, kb11;
    const bf16* kr0 = qkv + (size_t)(b * T_ + k0 + ln) * QKVD + D_ + h * HD_ + grp * 8;
    load16(kb00, kr0);
    load16(kb01, kr0 + 32);
    const bf16* kr1 = kr0 + (size_t)16 * QKVD;
    load16(kb10, kr1);
    load16(kb11, kr1 + 32);

    v8f z = {};
    v8f s0 = wmma_bf16(qa1, kb01, wmma_bf16(qa0, kb00, z));
    v8f s1 = wmma_bf16(qa1, kb11, wmma_bf16(qa0, kb10, z));

    const int kg0 = k0 + ln, kg1 = k0 + 16 + ln;
    const bool mk0 = mask[b * T_ + kg0] != 0;
    const bool mk1 = mask[b * T_ + kg1] != 0;

    float rmax[8];
#pragma unroll
    for (int v = 0; v < 8; ++v) {
      const int qg = qbase + v + grp * 8;
      float e0 = (mq[v] && mk0 && (kg0 <= qg)) ? s0[v] * scale : -1e30f;
      float e1 = (mq[v] && mk1 && (kg1 <= qg)) ? s1[v] * scale : -1e30f;
      s0[v] = e0; s1[v] = e1;
      rmax[v] = fmaxf(e0, e1);
    }
    // Row max across the 16 lanes holding each row
#pragma unroll
    for (int off = 1; off < 16; off <<= 1) {
#pragma unroll
      for (int v = 0; v < 8; ++v) rmax[v] = fmaxf(rmax[v], __shfl_xor(rmax[v], off, 32));
    }

    float rsum[8];
#pragma unroll
    for (int v = 0; v < 8; ++v) {
      const float mnew = fmaxf(mrow[v], rmax[v]);
      const float alpha = __expf(mrow[v] - mnew);
      mrow[v] = mnew;
      float p0 = (s0[v] < -1e29f) ? 0.0f : __expf(s0[v] - mnew);
      float p1 = (s1[v] < -1e29f) ? 0.0f : __expf(s1[v] - mnew);
      rsum[v] = p0 + p1;
      lrow[v] *= alpha;
#pragma unroll
      for (int j = 0; j < 4; ++j) o[j][v] *= alpha;
      sP[v + grp * 8][ln]      = (bf16)p0;
      sP[v + grp * 8][16 + ln] = (bf16)p1;
    }
#pragma unroll
    for (int off = 1; off < 16; off <<= 1) {
#pragma unroll
      for (int v = 0; v < 8; ++v) rsum[v] += __shfl_xor(rsum[v], off, 32);
    }
#pragma unroll
    for (int v = 0; v < 8; ++v) lrow[v] += rsum[v];

    __builtin_amdgcn_s_wait_tensorcnt(0);  // V tile landed in sV
    __syncthreads();                       // sP stores visible

    // O += P V : P as A-frag (16x32) from LDS; V as B-frags gathered from LDS.
    BF16Frag pa;
    loadA(pa, &sP[ln][0], grp * 8);
#pragma unroll
    for (int j = 0; j < 4; ++j) {
      BF16Frag vb;
#pragma unroll
      for (int i = 0; i < 16; ++i) vb.h[i] = sV[grp * 8 + (i & 7) + ((i >> 3) << 4)][j * 16 + ln];
      o[j] = wmma_bf16(pa, vb, o[j]);
    }
    __syncthreads();  // DScnt=0: sV reads done before next chunk's TDM overwrite
  }

  // Normalize and write bf16 context [b*T+t, D]
#pragma unroll
  for (int v = 0; v < 8; ++v) {
    const float inv = (lrow[v] > 0.0f) ? (1.0f / lrow[v]) : 0.0f;
    const int row = b * T_ + qbase + v + grp * 8;
#pragma unroll
    for (int j = 0; j < 4; ++j) {
      ctx[(size_t)row * D_ + h * HD_ + j * 16 + ln] = (bf16)(o[j][v] * inv);
    }
  }
}

// ---------------------------------------------------------------- launcher
extern "C" void kernel_launch(void* const* d_in, const int* in_sizes, int n_in,
                              void* d_out, int out_size, void* d_ws, size_t ws_size,
                              hipStream_t stream) {
  (void)in_sizes; (void)n_in; (void)out_size; (void)ws_size;

  const float*         x     = (const float*)d_in[0];
  const unsigned char* maskp = (const unsigned char*)d_in[1];  // bool [b*t]
  const float*         W_in  = (const float*)d_in[2];
  const float*         W_out = (const float*)d_in[3];

  const int NX    = B_ * T_ * D_;      // 4194304
  const int NWIN  = 3 * D_ * D_;       // 3145728
  const int NWOUT = D_ * D_;           // 1048576
  const size_t NQKV = (size_t)B_ * T_ * 3 * D_;
  // bf16 workspace layout (48 MiB total)
  bf16* xb    = (bf16*)d_ws;
  bf16* winb  = xb + NX;
  bf16* woutb = winb + NWIN;
  bf16* qkvb  = woutb + NWOUT;
  bf16* ctxb  = qkvb + NQKV;

  cvt_f32_bf16<<<(NX    + 255) / 256, 256, 0, stream>>>(x,     xb,    NX);
  cvt_f32_bf16<<<(NWIN  + 255) / 256, 256, 0, stream>>>(W_in,  winb,  NWIN);
  cvt_f32_bf16<<<(NWOUT + 255) / 256, 256, 0, stream>>>(W_out, woutb, NWOUT);

  // qkv = x @ W_in^T : [4096, 3072]
  gemm_bf16_nt<<<dim3((3 * D_) / 64, (B_ * T_) / 256), 256, 0, stream>>>(
      xb, winb, B_ * T_, 3 * D_, D_, nullptr, qkvb, nullptr);

  // fused causal attention -> ctx [4096, 1024] bf16
  attn_kernel<<<dim3(T_ / QT_, H_, B_), 32, 0, stream>>>(qkvb, maskp, ctxb);

  // out = ctx @ W_out^T with padding-mask zeroing fused : [4096, 1024] f32
  gemm_bf16_nt<<<dim3(D_ / 64, (B_ * T_) / 256), 256, 0, stream>>>(
      ctxb, woutb, B_ * T_, D_, D_, (float*)d_out, nullptr, maskp);
}
